// GRU_91311004713251
// MI455X (gfx1250) — matrix operand order repaired and checked by reading, not compiled, not run on hardware
//
#include <hip/hip_runtime.h>

typedef __attribute__((ext_vector_type(16))) _Float16 v16h;
typedef __attribute__((ext_vector_type(8)))  float    v8f;

#define BB 256
#define TT 2048
#define HH 100
#define KP 128      // K (=H) padded to 4 chunks of 32
#define NP 304      // 3H padded to 19 tiles of 16
#define HGS 304

// fast sigmoid: v_exp_f32 + v_rcp_f32 (saturates correctly at +/-inf)
__device__ __forceinline__ float fast_sigmoid(float x) {
    return __builtin_amdgcn_rcpf(1.0f + __expf(-x));
}
// fast tanh: 1 - 2/(exp(2x)+1); exp overflow -> rcp -> 0 -> +/-1
__device__ __forceinline__ float fast_tanh(float x) {
    return 1.0f - 2.0f * __builtin_amdgcn_rcpf(__expf(2.0f * x) + 1.0f);
}

__global__ __launch_bounds__(256)
void gru_cdna5_kernel(const float* __restrict__ x_i,  const float* __restrict__ v,
                      const float* __restrict__ w_ih, const float* __restrict__ w_hh,
                      const float* __restrict__ b_ih, const float* __restrict__ b_hh,
                      const float* __restrict__ w_out,const float* __restrict__ b_out,
                      float* __restrict__ out)
{
    __shared__ __align__(16) _Float16 whh16[NP*KP];   // B-operand, [n][k] padded
    __shared__ __align__(16) _Float16 h16[16*KP];     // A-operand, [m][k] padded
    __shared__ float h32[16*HH];                      // fp32 master state
    __shared__ float hg[16*HGS];                      // gate preactivations
    __shared__ float s_wih[3*HH*2];
    __shared__ float s_bih[3*HH];
    __shared__ float s_bhh[3*HH];
    __shared__ float s_wout[2*HH];
    __shared__ float s_bout[2];
    __shared__ float s_A[4];
    __shared__ float vbuf[16*2];

    const int tid   = threadIdx.x;
    const int wave  = tid >> 5;
    const int lane  = tid & 31;
    const int half  = lane >> 4;
    const int lm    = lane & 15;
    const int bbase = blockIdx.x * 16;

    // ---------------- one-time setup ----------------
    for (int i = tid; i < 3*HH*2; i += 256) s_wih[i] = w_ih[i];
    for (int i = tid; i < 3*HH;   i += 256) { s_bih[i] = b_ih[i]; s_bhh[i] = b_hh[i]; }
    for (int i = tid; i < 2*HH;   i += 256) s_wout[i] = w_out[i];
    if (tid < 2) s_bout[tid] = b_out[tid];

    // w_hh [3H,H] -> LDS f16 [n][k] with zero padding (B-operand source)
    for (int i = tid; i < NP*KP; i += 256) {
        int n = i >> 7, k = i & (KP-1);
        float val = (n < 3*HH && k < HH) ? w_hh[n*HH + k] : 0.0f;
        whh16[i] = (_Float16)val;
    }
    for (int i = tid; i < 16*KP; i += 256) h16[i] = (_Float16)0.0f;

    // A = w_out @ w_out^T  (2x2)
    if (tid < 4) {
        int i0 = tid >> 1, j0 = tid & 1;
        float s = 0.0f;
        for (int c = 0; c < HH; ++c) s += w_out[i0*HH + c] * w_out[j0*HH + c];
        s_A[tid] = s;
    }
    __syncthreads();

    // h0 = w_out^T A^{-1} (x_i - b_out)   (closed-form 2x2 solve per element)
    {
        float a00 = s_A[0], a01 = s_A[1], a11 = s_A[3];
        float inv_det = 1.0f / (a00*a11 - a01*a01);
        #pragma unroll
        for (int e = tid; e < 16*KP; e += 256) {
            int row = e >> 7, col = e & (KP-1);
            if (col < HH) {
                float r0 = x_i[(bbase+row)*2 + 0] - s_bout[0];
                float r1 = x_i[(bbase+row)*2 + 1] - s_bout[1];
                float c0 = (a11*r0 - a01*r1) * inv_det;
                float c1 = (a00*r1 - a01*r0) * inv_det;
                float h  = c0*s_wout[col] + c1*s_wout[HH + col];
                h32[row*HH + col] = h;
                h16[row*KP + col] = (_Float16)h;
            }
        }
        if (tid < 32) {
            int row = tid >> 1, d = tid & 1;
            vbuf[row*2 + d] = v[((size_t)(bbase+row)*TT + 0)*2 + d];
        }
    }
    __syncthreads();

    // ---------------- sequential scan ----------------
    union Frag { v16h v; uint4 q[2]; };

    for (int t = 0; t < TT; ++t) {
        // ---- hg = h @ w_hh^T  via WMMA f16->f32 ----
        // A frags: lanes 0-15 K{0-7,16-23}, lanes 16-31 K{8-15,24-31} per 32-K chunk
        Frag a0,a1,a2,a3;
        {
            const uint4* hp = (const uint4*)h16;            // 8 halves per uint4
            int rowb = lm * (KP/8);
            a0.q[0] = hp[rowb + 0*4 + half]; a0.q[1] = hp[rowb + 0*4 + 2 + half];
            a1.q[0] = hp[rowb + 1*4 + half]; a1.q[1] = hp[rowb + 1*4 + 2 + half];
            a2.q[0] = hp[rowb + 2*4 + half]; a2.q[1] = hp[rowb + 2*4 + 2 + half];
            a3.q[0] = hp[rowb + 3*4 + half]; a3.q[1] = hp[rowb + 3*4 + 2 + half];
        }
        for (int nt = wave; nt < 19; nt += 8) {
            // B frags: lanes 0-15 K{0-7,16-23}, lanes 16-31 K{8-15,24-31} per 32-K chunk
            const uint4* wp = (const uint4*)whh16;
            int colb = (nt*16 + lm) * (KP/8);
            Frag b0,b1,b2,b3;
            b0.q[0] = wp[colb + 0*4 + half]; b0.q[1] = wp[colb + 0*4 + 2 + half];
            b1.q[0] = wp[colb + 1*4 + half]; b1.q[1] = wp[colb + 1*4 + 2 + half];
            b2.q[0] = wp[colb + 2*4 + half]; b2.q[1] = wp[colb + 2*4 + 2 + half];
            b3.q[0] = wp[colb + 3*4 + half]; b3.q[1] = wp[colb + 3*4 + 2 + half];
            v8f acc = {};
            acc = __builtin_amdgcn_wmma_f32_16x16x32_f16(false, a0.v, false, b0.v, (short)0, acc, false, false);
            acc = __builtin_amdgcn_wmma_f32_16x16x32_f16(false, a1.v, false, b1.v, (short)0, acc, false, false);
            acc = __builtin_amdgcn_wmma_f32_16x16x32_f16(false, a2.v, false, b2.v, (short)0, acc, false, false);
            acc = __builtin_amdgcn_wmma_f32_16x16x32_f16(false, a3.v, false, b3.v, (short)0, acc, false, false);
            int cbase = nt*16 + lm;
            #pragma unroll
            for (int r = 0; r < 8; ++r)
                hg[(r + half*8)*HGS + cbase] = acc[r];
        }
        __syncthreads();

        // ---- gates + state update (fp32, hardware-rate transcendentals) ----
        // iterate padded space: row = e>>7, col = e&127 -> no integer division
        #pragma unroll
        for (int e = tid; e < 16*KP; e += 256) {
            int row = e >> 7, col = e & (KP-1);
            if (col < HH) {
                float v0 = vbuf[row*2+0], v1 = vbuf[row*2+1];
                float xr = v0*s_wih[col*2+0]        + v1*s_wih[col*2+1]        + s_bih[col];
                float xz = v0*s_wih[(HH+col)*2+0]   + v1*s_wih[(HH+col)*2+1]   + s_bih[HH+col];
                float xn = v0*s_wih[(2*HH+col)*2+0] + v1*s_wih[(2*HH+col)*2+1] + s_bih[2*HH+col];
                float hr = hg[row*HGS + col]        + s_bhh[col];
                float hz = hg[row*HGS + HH + col]   + s_bhh[HH+col];
                float hn = hg[row*HGS + 2*HH + col] + s_bhh[2*HH+col];
                float r = fast_sigmoid(xr + hr);
                float z = fast_sigmoid(xz + hz);
                float n = fast_tanh(xn + r*hn);
                float hold = h32[row*HH + col];
                float hnew = (1.0f - z)*n + z*hold;
                h32[row*HH + col] = hnew;
                h16[row*KP + col] = (_Float16)hnew;
            }
        }
        __syncthreads();

        // ---- output projection (t): all 256 threads, 8 lanes per (row,o) pair;
        //      butterfly-reduce within the 8-lane group via shfl_xor ----
        {
            int pair = tid >> 3;          // 0..31 : (row, o)
            int sub  = tid & 7;           // position within 8-lane group
            int row  = pair >> 1, o = pair & 1;
            const float* hrp = &h32[row*HH];
            const float* wrp = &s_wout[o*HH];
            float s = 0.0f;
            for (int c = sub; c < HH; c += 8) s += hrp[c]*wrp[c];
            s += __shfl_xor(s, 1, 32);
            s += __shfl_xor(s, 2, 32);
            s += __shfl_xor(s, 4, 32);
            if (sub == 0)
                out[((size_t)(bbase+row)*TT + t)*2 + o] = s + s_bout[o];
            else if (sub == 1 && (t+1) < TT) {
                // prefetch v(t+1) into LDS; read by elementwise(t+1) after barrier
                int q = pair;             // 0..31
                int vr = q >> 1, d = q & 1;
                vbuf[vr*2 + d] = v[((size_t)(bbase+vr)*TT + (t+1))*2 + d];
                if ((t + 16) < TT)
                    __builtin_prefetch(&v[((size_t)(bbase+vr)*TT + (t+16))*2 + d], 0, 0);
            }
        }
    }
}

extern "C" void kernel_launch(void* const* d_in, const int* in_sizes, int n_in,
                              void* d_out, int out_size, void* d_ws, size_t ws_size,
                              hipStream_t stream) {
    (void)in_sizes; (void)n_in; (void)out_size; (void)d_ws; (void)ws_size;
    const float* x_i   = (const float*)d_in[0];
    const float* v     = (const float*)d_in[1];
    const float* w_ih  = (const float*)d_in[2];
    const float* w_hh  = (const float*)d_in[3];
    const float* b_ih  = (const float*)d_in[4];
    const float* b_hh  = (const float*)d_in[5];
    const float* w_out = (const float*)d_in[6];
    const float* b_out = (const float*)d_in[7];
    float* outp = (float*)d_out;

    gru_cdna5_kernel<<<dim3(BB/16), dim3(256), 0, stream>>>(
        x_i, v, w_ih, w_hh, b_ih, b_hh, w_out, b_out, outp);
}
